// GNO_34746285425229
// MI455X (gfx1250) — matrix operand order repaired and checked by reading, not compiled, not run on hardware
//
#include <hip/hip_runtime.h>
#include <hip/hip_bf16.h>

#define HID 128
#define LDS_STRIDE 136   // 272 bytes: 16B aligned, 68-dword column pitch -> conflict-free

typedef __attribute__((ext_vector_type(16))) __bf16 v16bf;
typedef __attribute__((ext_vector_type(8)))  float  v8f;

// ---------------------------------------------------------------------------
// Kernel 1: lift  h = x @ W_lift + b_lift   and zero the agg buffer.
// One thread per (node, channel). Memory bound; fmaf chain over K=3.
// ---------------------------------------------------------------------------
__global__ void k_lift_zero(const float* __restrict__ x,
                            const float* __restrict__ Wl,
                            const float* __restrict__ bl,
                            float* __restrict__ h,
                            float* __restrict__ agg,
                            int n_nodes) {
    int idx = blockIdx.x * blockDim.x + threadIdx.x;
    int total = n_nodes * HID;
    if (idx >= total) return;
    int node = idx >> 7;
    int c    = idx & 127;
    float x0 = x[node * 3 + 0];
    float x1 = x[node * 3 + 1];
    float x2 = x[node * 3 + 2];
    float v = fmaf(x0, Wl[c],
              fmaf(x1, Wl[HID + c],
              fmaf(x2, Wl[2 * HID + c], bl[c])));
    h[idx]   = v;
    agg[idx] = 0.0f;
}

// ---------------------------------------------------------------------------
// Kernel 2: scatter-add  agg[dst] += h[src]  (segment_sum).
// One wave per edge iteration: lane owns 4 channels (float4 gather +
// 4x global_atomic_add_f32, no-return form). agg/h are L2-resident (51MB each
// vs 192MB L2) so this is L2-atomic bound; grid-stride keeps waves saturated.
// ---------------------------------------------------------------------------
__global__ void k_scatter(const int* __restrict__ ei,
                          const float* __restrict__ h,
                          float* __restrict__ agg,
                          int n_edges) {
    const int lane   = threadIdx.x & 31;
    const int wave   = (blockIdx.x * blockDim.x + threadIdx.x) >> 5;
    const int nwaves = (gridDim.x * blockDim.x) >> 5;
    const int c4     = lane * 4;
    for (int e = wave; e < n_edges; e += nwaves) {
        int src = ei[e];
        int dst = ei[n_edges + e];
        if (e + nwaves < n_edges) {
            // gfx1250 global_prefetch_b8 on the next edge indices
            __builtin_prefetch(&ei[e + nwaves], 0, 0);
            __builtin_prefetch(&ei[n_edges + e + nwaves], 0, 0);
        }
        const float4 v = *(const float4*)(h + (size_t)src * HID + c4);
        float* a = agg + (size_t)dst * HID + c4;
        atomicAdd(a + 0, v.x);
        atomicAdd(a + 1, v.y);
        atomicAdd(a + 2, v.z);
        atomicAdd(a + 3, v.w);
    }
}

// ---------------------------------------------------------------------------
// Kernel 3 (fused): out = tanh(agg@W_rel + b_rel + h@W_root) @ W_proj + b_proj
// One wave per 16-node row tile; 8 waves / 256-thread block.
// Weights staged transposed in LDS as bf16; v_wmma_f32_16x16x32_bf16 with
// f32 accumulation; tanh + projection fused so h2 never touches memory.
// ---------------------------------------------------------------------------
__global__ __launch_bounds__(256)
void k_gemm_tanh_proj(const float* __restrict__ agg,
                      const float* __restrict__ h,
                      const float* __restrict__ Wrel,
                      const float* __restrict__ brel,
                      const float* __restrict__ Wroot,
                      const float* __restrict__ Wproj,
                      const float* __restrict__ bproj,
                      float* __restrict__ out,
                      int n_nodes, int n_tiles) {
    __shared__ __bf16 sRel [HID * LDS_STRIDE];   // sRel [n][k] = W_rel [k][n]
    __shared__ __bf16 sRoot[HID * LDS_STRIDE];   // sRoot[n][k] = W_root[k][n]

    // Stage transposed bf16 weights (coalesced global reads along n).
    for (int i = threadIdx.x; i < HID * HID; i += blockDim.x) {
        int k = i >> 7;
        int n = i & 127;
        sRel [n * LDS_STRIDE + k] = (__bf16)Wrel [i];
        sRoot[n * LDS_STRIDE + k] = (__bf16)Wroot[i];
    }
    __syncthreads();

    const int lane   = threadIdx.x & 31;
    const int waveId = threadIdx.x >> 5;
    const int tile   = blockIdx.x * 8 + waveId;
    if (tile >= n_tiles) return;          // wave-uniform: EXEC stays all-ones

    const int half = lane >> 4;           // K-half selector (A/B layouts)
    const int l16  = lane & 15;
    int row = tile * 16 + l16;
    if (row >= n_nodes) row = n_nodes - 1;   // clamp (stores guarded below)

    // --- A fragments for all 4 K-steps, both matrices -----------------------
    // 16-bit A 16x32 layout: lane<16 -> K {kb..kb+7, kb+16..kb+23},
    //                        lane>=16 -> same +8.  Two contiguous float8 reads.
    v16bf Aagg[4], Ah[4];
#pragma unroll
    for (int kk = 0; kk < 4; ++kk) {
        const int kb = kk * 32 + half * 8;
        const float* pa = agg + (size_t)row * HID + kb;
        const float* ph = h   + (size_t)row * HID + kb;
#pragma unroll
        for (int i = 0; i < 8; ++i) {
            Aagg[kk][i]     = (__bf16)pa[i];
            Aagg[kk][i + 8] = (__bf16)pa[i + 16];
            Ah[kk][i]       = (__bf16)ph[i];
            Ah[kk][i + 8]   = (__bf16)ph[i + 16];
        }
    }

    float pf[8];
#pragma unroll
    for (int r = 0; r < 8; ++r) pf[r] = 0.0f;

    // --- 8 N-tiles of 16 columns each --------------------------------------
#pragma unroll
    for (int nt = 0; nt < 8; ++nt) {
        const int col  = nt * 16 + l16;      // B layout: lane holds column N
        const float bz = brel[col];
        v8f acc;
#pragma unroll
        for (int r = 0; r < 8; ++r) acc[r] = bz;   // bias folded into C

#pragma unroll
        for (int kk = 0; kk < 4; ++kk) {
            // 16-bit B 32x16 layout: lane<16 -> K {kb..kb+7, kb+16..kb+23}, lane>=16 -> +8.
            const int ks = kk * 32 + half * 8;
            const __bf16* pr = &sRel [col * LDS_STRIDE + ks];
            const __bf16* pq = &sRoot[col * LDS_STRIDE + ks];
            v16bf Br, Bq;
#pragma unroll
            for (int i = 0; i < 16; ++i) { Br[i] = pr[i + (i >> 3) * 8]; Bq[i] = pq[i + (i >> 3) * 8]; }
            acc = __builtin_amdgcn_wmma_f32_16x16x32_bf16(
                      false, Aagg[kk], false, Br, (short)0, acc, false, false);
            acc = __builtin_amdgcn_wmma_f32_16x16x32_bf16(
                      false, Ah[kk],   false, Bq, (short)0, acc, false, false);
        }

        // Fused tanh + projection partials: acc[r] is element (m = half*8+r,
        // n = col); each lane accumulates tanh(v)*W_proj[n] per m.
        const float wp = Wproj[col];
#pragma unroll
        for (int r = 0; r < 8; ++r)
            pf[r] = fmaf(tanhf(acc[r]), wp, pf[r]);
    }

    // Reduce over the 16 N-lanes of each half (xor 1,2,4,8 stays in-half).
#pragma unroll
    for (int m = 1; m < 16; m <<= 1) {
#pragma unroll
        for (int r = 0; r < 8; ++r)
            pf[r] += __shfl_xor(pf[r], m, 32);
    }

    if (l16 == 0) {
        const float bp   = bproj[0];
        const int  mbase = tile * 16 + half * 8;   // half0 -> M 0..7, half1 -> 8..15
#pragma unroll
        for (int r = 0; r < 8; ++r) {
            int m = mbase + r;
            if (m < n_nodes) out[m] = pf[r] + bp;
        }
    }
}

// ---------------------------------------------------------------------------
extern "C" void kernel_launch(void* const* d_in, const int* in_sizes, int n_in,
                              void* d_out, int out_size, void* d_ws, size_t ws_size,
                              hipStream_t stream) {
    const float* x     = (const float*)d_in[0];
    const int*   ei    = (const int*)  d_in[1];
    const float* Wl    = (const float*)d_in[2];
    const float* bl    = (const float*)d_in[3];
    const float* Wrel  = (const float*)d_in[4];
    const float* brel  = (const float*)d_in[5];
    const float* Wroot = (const float*)d_in[6];
    const float* Wproj = (const float*)d_in[7];
    const float* bproj = (const float*)d_in[8];
    float*       out   = (float*)d_out;

    const int n_nodes = in_sizes[0] / 3;
    const int n_edges = in_sizes[1] / 2;

    // Workspace: h [n_nodes*128] f32, agg [n_nodes*128] f32 (~102 MB total).
    float* h   = (float*)d_ws;
    float* agg = h + (size_t)n_nodes * HID;

    {
        int total  = n_nodes * HID;
        int blocks = (total + 255) / 256;
        k_lift_zero<<<blocks, 256, 0, stream>>>(x, Wl, bl, h, agg, n_nodes);
    }

    k_scatter<<<2048, 256, 0, stream>>>(ei, h, agg, n_edges);

    {
        int n_tiles = (n_nodes + 15) / 16;
        int blocks  = (n_tiles + 7) / 8;   // 8 waves (tiles) per block
        k_gemm_tanh_proj<<<blocks, 256, 0, stream>>>(
            agg, h, Wrel, brel, Wroot, Wproj, bproj, out, n_nodes, n_tiles);
    }
}
